// ViGBlock_72241349918942
// MI455X (gfx1250) — matrix operand order repaired and checked by reading, not compiled, not run on hardware
//
#include <hip/hip_runtime.h>
#include <hip/hip_bf16.h>
#include <math.h>

// ---------------- CDNA5 WMMA types ----------------
typedef __bf16 bf16x16 __attribute__((ext_vector_type(16)));
typedef float  f32x8   __attribute__((ext_vector_type(8)));

union Frag16 { bf16x16 v; int4 q[2]; };

__device__ __forceinline__ f32x8 zero8() {
    f32x8 z;
#pragma unroll
    for (int i = 0; i < 8; ++i) z[i] = 0.0f;
    return z;
}

__device__ __forceinline__ f32x8 wmma_bf16(const bf16x16& a, const bf16x16& b, const f32x8& c) {
    return __builtin_amdgcn_wmma_f32_16x16x32_bf16(false, a, false, b, (short)0, c, false, false);
}

// Problem constants
constexpr int Bc = 8;
constexpr int Cc = 192;
constexpr int Nc = 3136;   // 56*56, = 196*16, = 49*64
constexpr int Kn = 9;      // k*dilation, dilation=1

// ---------------- prep kernels ----------------
__global__ __launch_bounds__(256) void cvt_f32_bf16(const float* __restrict__ s,
                                                    __bf16* __restrict__ d, int n) {
    int i = blockIdx.x * 256 + threadIdx.x;
    if (i < n) d[i] = (__bf16)s[i];
}

// fold conv bias + BN into per-out-channel scale/shift: y = (Wx)*s + (b*s+sh)
__global__ __launch_bounds__(256) void make_st(const float* __restrict__ bias,
                                               const float* __restrict__ sc,
                                               const float* __restrict__ sh,
                                               float* __restrict__ s_out,
                                               float* __restrict__ t_out, int n) {
    int i = blockIdx.x * 256 + threadIdx.x;
    if (i < n) { s_out[i] = sc[i]; t_out[i] = bias[i] * sc[i] + sh[i]; }
}

// x [B,C,N] f32 -> xT [B,N,C] bf16 (token-major)
__global__ __launch_bounds__(256) void pack_x(const float* __restrict__ X,
                                              __bf16* __restrict__ XT) {
    long long tid = (long long)blockIdx.x * 256 + threadIdx.x;
    long long total = (long long)Bc * Nc * Cc;
    if (tid >= total) return;
    int c = (int)(tid % Cc);
    long long bn = tid / Cc;
    int n = (int)(bn % Nc);
    int b = (int)(bn / Nc);
    XT[tid] = (__bf16)X[((long long)b * Cc + c) * Nc + n];
}

// ---------------- fused 1x1-conv GEMM (bf16 WMMA, f32 acc) ----------------
// out[b,o,n] = act( (sum_i W[o,i]*in[b,n,i]) * s[o] + t[o] ) (+ resid[b,o,n])
// W: bf16 [O,I] row-major. X: bf16 [B,N,I] token-major.
// Each wave computes one 16(M) x 64(N) strip: 4 accumulators, A reused.
// K-loop is 2x unrolled with explicit double buffering so b128 fragment loads
// for kb+32 are in flight while the WMMA chain for kb executes.
template <bool GELU, bool RES, bool OUTF, bool OUTB>
__global__ __launch_bounds__(256) void gemm_bn(const __bf16* __restrict__ W,
                                               const __bf16* __restrict__ X,
                                               const float* __restrict__ s,
                                               const float* __restrict__ t,
                                               const float* __restrict__ resid, // [B,O,N]
                                               float* __restrict__ outf,        // [B,O,N]
                                               __bf16* __restrict__ outb,       // [B,N,O]
                                               int O, int I) {
    const int wave = threadIdx.x >> 5;
    const int lane = threadIdx.x & 31;
    const int l15  = lane & 15;
    const bool hi  = lane >= 16;

    const int mtiles = O / 16;
    const int nstrip = Nc / 64;
    int gw = blockIdx.x * 8 + wave;
    int total = Bc * mtiles * nstrip;
    if (gw >= total) return;

    int b  = gw / (mtiles * nstrip);
    int r  = gw % (mtiles * nstrip);
    int mt = r / nstrip;
    int ns = r % nstrip;
    const int mbase = mt * 16;
    const int nbase = ns * 64;

    const __bf16* Wrow  = W + (long long)(mbase + l15) * I;
    const __bf16* Xbase = X + ((long long)b * Nc + nbase) * I;

    f32x8 acc0 = zero8(), acc1 = zero8(), acc2 = zero8(), acc3 = zero8();
    const int hiA = hi ? 8 : 0;
    const int hiB = hi ? 8 : 0;

    auto loadA = [&](int kb, Frag16& a) {
        a.q[0] = *reinterpret_cast<const int4*>(Wrow + kb + hiA);
        a.q[1] = *reinterpret_cast<const int4*>(Wrow + kb + 16 + hiA);
    };
    auto loadB4 = [&](int kb, Frag16* bf) {
#pragma unroll
        for (int nt = 0; nt < 4; ++nt) {
            const __bf16* xr = Xbase + (long long)(nt * 16 + l15) * I + kb + hiB;
            bf[nt].q[0] = *reinterpret_cast<const int4*>(xr);
            bf[nt].q[1] = *reinterpret_cast<const int4*>(xr + 16);
        }
    };
    auto mma4 = [&](const Frag16& a, const Frag16* bf) {
        acc0 = wmma_bf16(a.v, bf[0].v, acc0);
        acc1 = wmma_bf16(a.v, bf[1].v, acc1);
        acc2 = wmma_bf16(a.v, bf[2].v, acc2);
        acc3 = wmma_bf16(a.v, bf[3].v, acc3);
    };

    Frag16 a0, a1, bb0[4], bb1[4];
    loadA(0, a0);
    loadB4(0, bb0);
    // I/32 is always even (6, 12 or 24 K-blocks): process two K-blocks per trip
    for (int kb = 0; kb < I; kb += 64) {
        loadA(kb + 32, a1);
        loadB4(kb + 32, bb1);
        mma4(a0, bb0);
        if (kb + 64 < I) {
            loadA(kb + 64, a0);
            loadB4(kb + 64, bb0);
        }
        mma4(a1, bb1);
    }

    // epilogue: D-frag lane mapping -> (m = mbase + v + hi*8, n = nbase + nt*16 + l15)
    f32x8 accs[4] = {acc0, acc1, acc2, acc3};
#pragma unroll
    for (int nt = 0; nt < 4; ++nt) {
        int n = nbase + nt * 16 + l15;
#pragma unroll
        for (int v = 0; v < 8; ++v) {
            int m = mbase + v + (hi ? 8 : 0);
            float val = accs[nt][v] * s[m] + t[m];
            if (GELU) val = 0.5f * val * (1.0f + erff(val * 0.70710678118654752f));
            if (RES) val += resid[((long long)b * O + m) * Nc + n];
            if (OUTF) outf[((long long)b * O + m) * Nc + n] = val;
            if (OUTB) outb[((long long)b * Nc + n) * O + m] = (__bf16)val;
        }
    }
}

// ---------------- per-token L2 normalization (wave per token) ----------------
__global__ __launch_bounds__(256) void norm_tokens(const __bf16* __restrict__ H,
                                                   __bf16* __restrict__ HN) {
    int wave = threadIdx.x >> 5;
    int lane = threadIdx.x & 31;
    int token = blockIdx.x * 8 + wave;
    if (token >= Bc * Nc) return;
    const __bf16* hp = H + (long long)token * Cc;
    float ss = 0.0f;
#pragma unroll
    for (int c = lane; c < Cc; c += 32) { float v = (float)hp[c]; ss += v * v; }
#pragma unroll
    for (int m = 16; m > 0; m >>= 1) ss += __shfl_xor(ss, m, 32);
    float r = rsqrtf(ss + 1e-12f);
#pragma unroll
    for (int c = lane; c < Cc; c += 32)
        HN[(long long)token * Cc + c] = (__bf16)((float)hp[c] * r);
}

// ---------------- fused similarity + top-9 kNN (WMMA gram matrix) ----------------
// dist = 2 - 2*sim for unit vectors -> top-k(-dist) == top-k(sim). Wave handles
// 16 query rows; iterates all 196 column tiles with ping-pong B-fragment
// buffers so the next tile's 12 b128 loads overlap the current WMMA chain and
// top-k scan. D tile bounces via per-wave LDS (wave-internal DS in-order);
// lanes 0..15 keep a sorted top-9 (value,index) in registers.
__global__ __launch_bounds__(256) void knn_topk(const __bf16* __restrict__ HN,
                                                int* __restrict__ IDX) {
    __shared__ float sm[8][16][17];
    const int wave = threadIdx.x >> 5;
    const int lane = threadIdx.x & 31;
    const int l15  = lane & 15;
    const bool hi  = lane >= 16;

    const int rtiles = Nc / 16;
    int gw = blockIdx.x * 8 + wave;
    if (gw >= Bc * rtiles) return;
    int b  = gw / rtiles;
    int rt = gw % rtiles;

    const int hiA = hi ? 8 : 0;
    const int hiB = hi ? 8 : 0;
    const __bf16* base = HN + (long long)b * Nc * Cc;

    // preload all 6 K-block A fragments for the 16 query rows
    Frag16 A[6];
    const __bf16* Arow = base + (long long)(rt * 16 + l15) * Cc;
#pragma unroll
    for (int kb = 0; kb < 6; ++kb) {
        A[kb].q[0] = *reinterpret_cast<const int4*>(Arow + kb * 32 + hiA);
        A[kb].q[1] = *reinterpret_cast<const int4*>(Arow + kb * 32 + 16 + hiA);
    }

    float tv[Kn]; int ti[Kn];
#pragma unroll
    for (int j = 0; j < Kn; ++j) { tv[j] = -3.0e38f; ti[j] = 0; }

    auto loadB = [&](int ct, Frag16* dst) {
        const __bf16* Brow = base + (long long)(ct * 16 + l15) * Cc + hiB;
#pragma unroll
        for (int kb = 0; kb < 6; ++kb) {
            dst[kb].q[0] = *reinterpret_cast<const int4*>(Brow + kb * 32);
            dst[kb].q[1] = *reinterpret_cast<const int4*>(Brow + kb * 32 + 16);
        }
    };
    auto process = [&](int ct, const Frag16* Bf) {
        f32x8 acc = zero8();
#pragma unroll
        for (int kb = 0; kb < 6; ++kb) acc = wmma_bf16(A[kb].v, Bf[kb].v, acc);
#pragma unroll
        for (int v = 0; v < 8; ++v) sm[wave][v + (hi ? 8 : 0)][l15] = acc[v];
        __builtin_amdgcn_wave_barrier();
        if (!hi) {
#pragma unroll
            for (int j = 0; j < 16; ++j) {
                float vsc = sm[wave][l15][j];
                int gi = ct * 16 + j;
                if (vsc > tv[Kn - 1]) {
                    tv[Kn - 1] = vsc; ti[Kn - 1] = gi;
#pragma unroll
                    for (int q = Kn - 1; q > 0; --q) {
                        if (tv[q] > tv[q - 1]) {
                            float tf = tv[q]; tv[q] = tv[q - 1]; tv[q - 1] = tf;
                            int   td = ti[q]; ti[q] = ti[q - 1]; ti[q - 1] = td;
                        }
                    }
                }
            }
        }
        __builtin_amdgcn_wave_barrier();
    };

    Frag16 B0[6], B1[6];
    loadB(0, B0);
    // rtiles = 196 (even): ping-pong two column tiles per trip
    for (int ct = 0; ct < rtiles; ct += 2) {
        loadB(ct + 1, B1);
        process(ct, B0);
        if (ct + 2 < rtiles) loadB(ct + 2, B0);
        process(ct + 1, B1);
    }

    if (!hi) {
        int row = rt * 16 + l15;
        int* op = IDX + ((long long)b * Nc + row) * Kn;
#pragma unroll
        for (int j = 0; j < Kn; ++j) op[j] = ti[j];
    }
}

// ---------------- max-relative aggregation + concat ----------------
// cat[b,n,0:C] = h ; cat[b,n,C:2C] = max_j (h[idx_j] - h)
__global__ __launch_bounds__(256) void maxrel_cat(const __bf16* __restrict__ H,
                                                  const int* __restrict__ IDX,
                                                  __bf16* __restrict__ CAT) {
    long long tid = (long long)blockIdx.x * 256 + threadIdx.x;
    long long total = (long long)Bc * Nc * Cc;
    if (tid >= total) return;
    int c = (int)(tid % Cc);
    long long bn = tid / Cc;
    int b = (int)(bn / Nc);
    float hb = (float)H[bn * Cc + c];
    const int* ip = IDX + bn * Kn;
    float mx = -3.0e38f;
#pragma unroll
    for (int j = 0; j < Kn; ++j) {
        int nj = ip[j];
        float hj = (float)H[((long long)b * Nc + nj) * Cc + c];
        mx = fmaxf(mx, hj - hb);
    }
    CAT[bn * (2 * Cc) + c] = (__bf16)hb;
    CAT[bn * (2 * Cc) + Cc + c] = (__bf16)mx;
}

// ---------------- host orchestration ----------------
extern "C" void kernel_launch(void* const* d_in, const int* in_sizes, int n_in,
                              void* d_out, int out_size, void* d_ws, size_t ws_size,
                              hipStream_t stream) {
    const float* x     = (const float*)d_in[0];
    const float* g1_w  = (const float*)d_in[1];
    const float* g1_b  = (const float*)d_in[2];
    const float* g1_s  = (const float*)d_in[3];
    const float* g1_sh = (const float*)d_in[4];
    const float* mr_w  = (const float*)d_in[5];
    const float* mr_b  = (const float*)d_in[6];
    const float* gc_s  = (const float*)d_in[7];
    const float* gc_sh = (const float*)d_in[8];
    const float* g2_w  = (const float*)d_in[9];
    const float* g2_b  = (const float*)d_in[10];
    const float* g2_s  = (const float*)d_in[11];
    const float* g2_sh = (const float*)d_in[12];
    const float* f1_w  = (const float*)d_in[13];
    const float* f1_b  = (const float*)d_in[14];
    const float* f1_s  = (const float*)d_in[15];
    const float* f1_sh = (const float*)d_in[16];
    const float* f2_w  = (const float*)d_in[17];
    const float* f2_b  = (const float*)d_in[18];
    const float* f2_s  = (const float*)d_in[19];
    const float* f2_sh = (const float*)d_in[20];
    float* out = (float*)d_out;

    // bump allocator over workspace (sizes are multiples of 256B by construction)
    char* wsp = (char*)d_ws;
    auto alloc = [&](size_t bytes) -> char* {
        char* p = wsp;
        wsp += (bytes + 255) & ~(size_t)255;
        return p;
    };
    const size_t BNC = (size_t)Bc * Nc * Cc;

    __bf16* wg1 = (__bf16*)alloc(192 * 192 * 2);
    __bf16* wmr = (__bf16*)alloc(384 * 384 * 2);
    __bf16* wg2 = (__bf16*)alloc(192 * 384 * 2);
    __bf16* wf1 = (__bf16*)alloc(768 * 192 * 2);
    __bf16* wf2 = (__bf16*)alloc(192 * 768 * 2);
    float* s_g1 = (float*)alloc(192 * 4);  float* t_g1 = (float*)alloc(192 * 4);
    float* s_gc = (float*)alloc(384 * 4);  float* t_gc = (float*)alloc(384 * 4);
    float* s_g2 = (float*)alloc(192 * 4);  float* t_g2 = (float*)alloc(192 * 4);
    float* s_f1 = (float*)alloc(768 * 4);  float* t_f1 = (float*)alloc(768 * 4);
    float* s_f2 = (float*)alloc(192 * 4);  float* t_f2 = (float*)alloc(192 * 4);

    __bf16* xT  = (__bf16*)alloc(BNC * 2);          // [B,N,C]; reused later as xg_bf16
    __bf16* h   = (__bf16*)alloc(BNC * 2);          // [B,N,C]   (contiguous h|hn|g)
    __bf16* hn  = (__bf16*)alloc(BNC * 2);          // [B,N,C]
    __bf16* g   = (__bf16*)alloc(BNC * 2 * 2);      // [B,N,2C]
    int*    idx = (int*)alloc((size_t)Bc * Nc * Kn * 4);
    __bf16* cat = (__bf16*)alloc(BNC * 2 * 2);      // [B,N,2C]; reused as xg_f32 [B,C,N]

    __bf16* f       = h;            // [B,N,4C] overlays dead h|hn|g (exact size match)
    __bf16* xg_bf16 = xT;           // xT dead after g1
    float*  xg_f32  = (float*)cat;  // cat dead after mr GEMM

    (void)in_sizes; (void)n_in; (void)out_size; (void)ws_size;

    auto gb = [](long long n) { return dim3((unsigned)((n + 255) / 256)); };
    dim3 blk(256);

    // ---- prep: convert weights, fold BN, pack x ----
    cvt_f32_bf16<<<gb(192 * 192), blk, 0, stream>>>(g1_w, wg1, 192 * 192);
    cvt_f32_bf16<<<gb(384 * 384), blk, 0, stream>>>(mr_w, wmr, 384 * 384);
    cvt_f32_bf16<<<gb(192 * 384), blk, 0, stream>>>(g2_w, wg2, 192 * 384);
    cvt_f32_bf16<<<gb(768 * 192), blk, 0, stream>>>(f1_w, wf1, 768 * 192);
    cvt_f32_bf16<<<gb(192 * 768), blk, 0, stream>>>(f2_w, wf2, 192 * 768);
    make_st<<<gb(192), blk, 0, stream>>>(g1_b, g1_s, g1_sh, s_g1, t_g1, 192);
    make_st<<<gb(384), blk, 0, stream>>>(mr_b, gc_s, gc_sh, s_gc, t_gc, 384);
    make_st<<<gb(192), blk, 0, stream>>>(g2_b, g2_s, g2_sh, s_g2, t_g2, 192);
    make_st<<<gb(768), blk, 0, stream>>>(f1_b, f1_s, f1_sh, s_f1, t_f1, 768);
    make_st<<<gb(192), blk, 0, stream>>>(f2_b, f2_s, f2_sh, s_f2, t_f2, 192);
    pack_x<<<gb((long long)Bc * Nc * Cc), blk, 0, stream>>>(x, xT);

    auto gemm_grid = [](int O) {
        int waves = Bc * (O / 16) * (Nc / 64);
        return dim3((unsigned)((waves + 7) / 8));
    };

    // ---- grapher fc1: h = BN(g1_w @ x) ----
    gemm_bn<false, false, false, true><<<gemm_grid(192), blk, 0, stream>>>(
        wg1, xT, s_g1, t_g1, nullptr, nullptr, h, 192, 192);

    // ---- kNN graph on normalized h ----
    norm_tokens<<<dim3((Bc * Nc + 7) / 8), blk, 0, stream>>>(h, hn);
    knn_topk<<<dim3((Bc * (Nc / 16) + 7) / 8), blk, 0, stream>>>(hn, idx);

    // ---- max-relative aggregation + concat ----
    maxrel_cat<<<gb((long long)Bc * Nc * Cc), blk, 0, stream>>>(h, idx, cat);

    // ---- mr: g = GELU(BN(mr_w @ cat)) ----
    gemm_bn<true, false, false, true><<<gemm_grid(384), blk, 0, stream>>>(
        wmr, cat, s_gc, t_gc, nullptr, nullptr, g, 384, 384);

    // ---- g2: xg = BN(g2_w @ g) + x  (write f32 channel-major + bf16 token-major) ----
    gemm_bn<false, true, true, true><<<gemm_grid(192), blk, 0, stream>>>(
        wg2, g, s_g2, t_g2, x, xg_f32, xg_bf16, 192, 384);

    // ---- ffn fc1: f = GELU(BN(f1_w @ xg)) ----
    gemm_bn<true, false, false, true><<<gemm_grid(768), blk, 0, stream>>>(
        wf1, xg_bf16, s_f1, t_f1, nullptr, nullptr, f, 768, 192);

    // ---- ffn fc2 + residual: out = BN(f2_w @ f) + xg ----
    gemm_bn<false, true, true, false><<<gemm_grid(192), blk, 0, stream>>>(
        wf2, f, s_f2, t_f2, xg_f32, out, nullptr, 192, 768);
}
